// VFInterpolator_13657996001995
// MI455X (gfx1250) — compile-verified, matrix operand order checked by reading, not run on hardware
//
#include <hip/hip_runtime.h>
#include <hip/hip_bf16.h>
#include <math.h>

typedef __bf16 bf16_t;
typedef bf16_t v16bf __attribute__((ext_vector_type(16)));
typedef float  v8f   __attribute__((ext_vector_type(8)));

#define DD_   128
#define HH_   256
#define GG_   50
#define NG_   8
#define LL_   3
#define EPS_  1e-5f

#define EDGE_WAVES 4
#define NODE_WAVES 2

#define GSTEP_  (10.0f / 49.0f)
#define GCOEFF_ (-0.5f / (GSTEP_ * GSTEP_))

// Packed-fragment sizes (frags of 32 lanes x 16 bf16 = 512 elements each).
#define FRAGS_EW1 16   // 8 ntiles * 2 ksteps  (K 50->64)
#define FRAGS_EW2 32   // 8 ntiles * 4 ksteps  (K 128)
#define FRAGS_NW1 64   // 16 ntiles * 4 ksteps (K 128)
#define FRAGS_NW2 64   // 8 ntiles * 8 ksteps  (K 256)

// ---------------------------------------------------------------- helpers ---

// Wave-private LDS producer->consumer sync (no cross-wave sharing, so a
// DScnt wait + scheduling fence is sufficient; avoids s_barrier entirely).
__device__ __forceinline__ void wave_sync_lds() {
  __builtin_amdgcn_sched_barrier(0);
  asm volatile("s_wait_dscnt 0x0" ::: "memory");
  __builtin_amdgcn_sched_barrier(0);
}

__device__ __forceinline__ v8f zero8() {
  v8f z;
#pragma unroll
  for (int i = 0; i < 8; ++i) z[i] = 0.f;
  return z;
}

__device__ __forceinline__ float silu_f(float x) { return x / (1.f + __expf(-x)); }

__device__ __forceinline__ v8f wmma_bf16(v16bf a, v16bf b, v8f c) {
  // (neg_a, A, neg_b, B, c_mod, C, reuse_a, reuse_b)
  return __builtin_amdgcn_wmma_f32_16x16x32_bf16(false, a, false, b, (short)0, c,
                                                 false, false);
}

// Inverse of the ISA 7.12.2 16-bit A/B fragment mapping: element (m, k%32) of a
// 16x32 tile lives at fragment lane frag_lane(), vector element frag_elem().
__device__ __forceinline__ int frag_lane(int m, int krem) {
  return m + ((krem & 8) ? 16 : 0);
}
__device__ __forceinline__ int frag_elem(int krem) {
  return (krem & 7) + ((krem & 16) ? 8 : 0);
}

// --------------------------------------------------------- weight packing ---
// Pre-swizzle a global f32 row-major weight matrix W[K][ldn] into fragment-linear
// bf16: frag index = ntile*nksteps + kstep; each fragment is 32 lanes x v16bf laid
// out exactly as v_wmma_*_bf16 wants its B operand.  K-padding handled here once.
__global__ __launch_bounds__(256) void pack_b_kernel(const float* __restrict__ W,
                                                     bf16_t* __restrict__ outp,
                                                     int Kvalid, int ldn,
                                                     int nksteps, int nfrags) {
  const int frag = blockIdx.x * 8 + (threadIdx.x >> 5);
  if (frag >= nfrags) return;
  const int lane = threadIdx.x & 31;
  const int nt = frag / nksteps;
  const int ks = frag % nksteps;
  const int n  = nt * 16 + (lane & 15);
  const int kb = (lane < 16) ? 0 : 8;
  v16bf f;
#pragma unroll
  for (int j = 0; j < 8; ++j) {
    const int k = ks * 32 + ((j < 4) ? (kb + 2 * j) : (kb + 2 * j + 8));
    f[2 * j]     = (bf16_t)((k     < Kvalid) ? W[k * ldn + n]       : 0.f);
    f[2 * j + 1] = (bf16_t)((k + 1 < Kvalid) ? W[(k + 1) * ldn + n] : 0.f);
  }
  ((v16bf*)outp)[frag * 32 + lane] = f;
}

// ------------------------------------------------------------- tiny kernels ---

__global__ __launch_bounds__(256) void count_kernel(const int* __restrict__ batch,
                                                    float* __restrict__ cnts, int N) {
  int n = blockIdx.x * 256 + threadIdx.x;
  if (n < N) atomicAdd(&cnts[batch[n]], 1.0f);
}

__global__ __launch_bounds__(256) void init_h_kernel(const float* __restrict__ emb,
                                                     const int* __restrict__ atype,
                                                     float* __restrict__ h, int N) {
  int idx = blockIdx.x * 256 + threadIdx.x;
  if (idx < N * DD_) {
    int n = idx >> 7, c = idx & (DD_ - 1);
    h[idx] = emb[atype[n] * DD_ + c];
  }
}

__global__ __launch_bounds__(256) void div_kernel(float* __restrict__ dstv,
                                                  const float* __restrict__ srcv,
                                                  const float* __restrict__ cnts) {
  int idx = blockIdx.x * 256 + threadIdx.x;
  if (idx < NG_ * DD_) dstv[idx] = srcv[idx] / cnts[idx >> 7];
}

__global__ __launch_bounds__(256) void center_kernel(float* __restrict__ zbuf,
                                                     const float* __restrict__ gmean,
                                                     const float* __restrict__ gms,
                                                     const int* __restrict__ batch,
                                                     float* __restrict__ gvar, int N) {
  int idx = blockIdx.x * 256 + threadIdx.x;
  if (idx < N * DD_) {
    int n = idx >> 7, c = idx & (DD_ - 1);
    int g = batch[n];
    float zc = zbuf[idx] - gmean[g * DD_ + c] * gms[c];
    zbuf[idx] = zc;
    atomicAdd(&gvar[g * DD_ + c], zc * zc);
  }
}

__global__ __launch_bounds__(256) void apply_kernel(float* __restrict__ h,
                                                    const float* __restrict__ zbuf,
                                                    const float* __restrict__ gvm,
                                                    const float* __restrict__ gw,
                                                    const float* __restrict__ gb,
                                                    const int* __restrict__ batch,
                                                    const float* __restrict__ alp,
                                                    int layer, int N) {
  int idx = blockIdx.x * 256 + threadIdx.x;
  if (idx < N * DD_) {
    int n = idx >> 7, c = idx & (DD_ - 1);
    int g = batch[n];
    float t = gw[c] * zbuf[idx] * rsqrtf(gvm[g * DD_ + c] + EPS_) + gb[c];
    h[idx] += alp[layer] * silu_f(t);
  }
}

__global__ __launch_bounds__(256) void pool_accum(const float* __restrict__ h,
                                                  const int* __restrict__ batch,
                                                  float* __restrict__ psum, int N) {
  int idx = blockIdx.x * 256 + threadIdx.x;
  if (idx < N * DD_) {
    int n = idx >> 7, c = idx & (DD_ - 1);
    atomicAdd(&psum[batch[n] * DD_ + c], h[idx]);
  }
}

__global__ __launch_bounds__(256) void pool_finish(const float* __restrict__ psum,
                                                   const float* __restrict__ cnts,
                                                   float* __restrict__ out) {
  int g = threadIdx.x >> 5;
  int lane = threadIdx.x & 31;
  float s = 0.f;
  for (int c = lane; c < DD_; c += 32) s += psum[g * DD_ + c];
#pragma unroll
  for (int m = 16; m >= 1; m >>= 1) s += __shfl_xor(s, m, 32);
  if (lane == 0) out[g] = s / (cnts[g] * (float)DD_);
}

// ------------------------------------------------------------- edge kernel ---
// Fused: dist -> Gaussian smear -> Linear(50,128) -> LN -> SiLU -> Linear(128,128)
//        -> msg = e * h[src] -> atomicAdd h0[dst].
// One wave handles 16 edges; GEMMs via v_wmma_f32_16x16x32_bf16 with A staged in
// LDS directly in fragment layout and B pre-packed in global bf16.
__global__ __launch_bounds__(32 * EDGE_WAVES) void edge_kernel(
    const float* __restrict__ pos, const int* __restrict__ src,
    const int* __restrict__ dstp, const float* __restrict__ h,
    float* __restrict__ h0,
    const bf16_t* __restrict__ pw1, const bf16_t* __restrict__ pw2,
    const float* __restrict__ eb1, const float* __restrict__ eg1,
    const float* __restrict__ ebe1, const float* __restrict__ eb2, int E) {
  __shared__ v16bf A0F[EDGE_WAVES][2][32];   // gaussian feats, fragment layout
  __shared__ float MF[EDGE_WAVES][16][128];  // GEMM1 f32 output
  __shared__ v16bf MBF[EDGE_WAVES][4][32];   // LN+SiLU bf16, fragment layout
  __shared__ int   SI[EDGE_WAVES][16];
  __shared__ int   DI[EDGE_WAVES][16];

  const int lane = threadIdx.x & 31;
  const int wv   = threadIdx.x >> 5;
  const int base = (blockIdx.x * EDGE_WAVES + wv) * 16;
  if (base >= E) return;
  const int ec = (E - base < 16) ? (E - base) : 16;

  const v16bf* B1 = (const v16bf*)pw1;
  const v16bf* B2 = (const v16bf*)pw2;
  __builtin_prefetch(pw1, 0, 1);
  __builtin_prefetch(pw2, 0, 1);

  // ---- Gaussian smearing: lanes (e, half) pair; 25 gaussians each half ----
  {
    const int e   = lane >> 1;
    const int hlf = lane & 1;
    const int ee  = (e < ec) ? e : (ec - 1);
    const int s = src[base + ee];
    const int d = dstp[base + ee];
    const float dx = pos[3 * s + 0] - pos[3 * d + 0];
    const float dy = pos[3 * s + 1] - pos[3 * d + 1];
    const float dz = pos[3 * s + 2] - pos[3 * d + 2];
    const float dist = sqrtf(dx * dx + dy * dy + dz * dz);
    if (!hlf) { SI[wv][e] = s; DI[wv][e] = d; }
    for (int g = hlf * 25; g < hlf * 25 + 25; ++g) {
      const float t = dist - (float)g * GSTEP_;
      bf16_t* fp = (bf16_t*)&A0F[wv][g >> 5][frag_lane(e, g & 31)];
      fp[frag_elem(g & 31)] = (bf16_t)__expf(GCOEFF_ * t * t);
    }
    if (hlf) {
      for (int g = GG_; g < 64; ++g) {
        bf16_t* fp = (bf16_t*)&A0F[wv][g >> 5][frag_lane(e, g & 31)];
        fp[frag_elem(g & 31)] = (bf16_t)0.f;
      }
    }
  }
  wave_sync_lds();

  const int mo = (lane < 16) ? 0 : 8;
  const int nn = lane & 15;

  // ---- GEMM1: [16x50] @ [50x128] + b1 -> MF ----
#pragma unroll
  for (int nt = 0; nt < 8; ++nt) {
    v8f acc = zero8();
#pragma unroll
    for (int ks = 0; ks < 2; ++ks) {
      v16bf a = A0F[wv][ks][lane];
      v16bf b = B1[(nt * 2 + ks) * 32 + lane];
      acc = wmma_bf16(a, b, acc);
    }
    const int c = nt * 16 + nn;
    const float bias = eb1[c];
#pragma unroll
    for (int r = 0; r < 8; ++r) MF[wv][mo + r][c] = acc[r] + bias;
  }
  wave_sync_lds();

  // ---- LayerNorm(128) + SiLU -> MBF (fragment layout) ----
  for (int row = 0; row < 16; ++row) {
    float v[4], s = 0.f, ss = 0.f;
#pragma unroll
    for (int j = 0; j < 4; ++j) {
      v[j] = MF[wv][row][lane * 4 + j];
      s += v[j];
      ss += v[j] * v[j];
    }
#pragma unroll
    for (int m = 16; m >= 1; m >>= 1) {
      s  += __shfl_xor(s, m, 32);
      ss += __shfl_xor(ss, m, 32);
    }
    const float mean = s * (1.f / 128.f);
    const float var  = ss * (1.f / 128.f) - mean * mean;
    const float rstd = rsqrtf(var + EPS_);
#pragma unroll
    for (int j = 0; j < 4; ++j) {
      const int c = lane * 4 + j;
      float t = eg1[c] * (v[j] - mean) * rstd + ebe1[c];
      bf16_t* fp = (bf16_t*)&MBF[wv][c >> 5][frag_lane(row, c & 31)];
      fp[frag_elem(c & 31)] = (bf16_t)silu_f(t);
    }
  }
  wave_sync_lds();

  // ---- GEMM2: [16x128] @ [128x128] + b2, fused message scatter ----
#pragma unroll
  for (int nt = 0; nt < 8; ++nt) {
    v8f acc = zero8();
#pragma unroll
    for (int ks = 0; ks < 4; ++ks) {
      v16bf a = MBF[wv][ks][lane];
      v16bf b = B2[(nt * 4 + ks) * 32 + lane];
      acc = wmma_bf16(a, b, acc);
    }
    const int c = nt * 16 + nn;
    const float bias = eb2[c];
    if (ec == 16) {
#pragma unroll
      for (int r = 0; r < 8; ++r) {
        const int er = mo + r;
        const float msg = (acc[r] + bias) * h[SI[wv][er] * DD_ + c];
        atomicAdd(&h0[DI[wv][er] * DD_ + c], msg);
      }
    } else {
#pragma unroll
      for (int r = 0; r < 8; ++r) {
        const int er = mo + r;
        if (er < ec) {
          const float msg = (acc[r] + bias) * h[SI[wv][er] * DD_ + c];
          atomicAdd(&h0[DI[wv][er] * DD_ + c], msg);
        }
      }
    }
  }
}

// ------------------------------------------------------------- node kernel ---
// z = conv_alpha*h0 + h; Linear(128,256) -> LN -> SiLU -> Linear(256,128)+b2.
// final layer: h += alpha*z.  else: z=SiLU(LN2(z)); zbuf=z; gsum[batch]+=z.
__global__ __launch_bounds__(32 * NODE_WAVES) void node_kernel(
    const float* hconst, const float* __restrict__ h0, float* hmut,
    const float* __restrict__ ca,
    const bf16_t* __restrict__ pw1, const bf16_t* __restrict__ pw2,
    const float* __restrict__ nb1, const float* __restrict__ ng1,
    const float* __restrict__ nbe1, const float* __restrict__ nb2,
    const float* __restrict__ ng2, const float* __restrict__ nbe2,
    const float* __restrict__ alp, int layer, int final_flag,
    float* __restrict__ zbuf, float* __restrict__ gsum,
    const int* __restrict__ batch, int Nn) {
  __shared__ v16bf A1F[NODE_WAVES][4][32];   // input acts, fragment layout
  __shared__ float MF[NODE_WAVES][16][256];  // GEMM1 f32 output (reused for z)
  __shared__ v16bf MBF[NODE_WAVES][8][32];   // LN+SiLU bf16, fragment layout

  const int lane = threadIdx.x & 31;
  const int wv   = threadIdx.x >> 5;
  const int base = (blockIdx.x * NODE_WAVES + wv) * 16;
  if (base >= Nn) return;
  const int rc = (Nn - base < 16) ? (Nn - base) : 16;

  const v16bf* B1 = (const v16bf*)pw1;
  const v16bf* B2 = (const v16bf*)pw2;
  __builtin_prefetch(pw1, 0, 1);
  __builtin_prefetch(pw2, 0, 1);

  // ---- stage A = bf16(conv_alpha*h0 + h), fragment layout ----
  {
    const float4 cav = *(const float4*)&ca[lane * 4];
    for (int row = 0; row < 16; ++row) {
      const int n = base + ((row < rc) ? row : (rc - 1));
      const float4 hv  = *(const float4*)&hconst[n * DD_ + lane * 4];
      const float4 h0v = *(const float4*)&h0[n * DD_ + lane * 4];
      float av[4] = {cav.x * h0v.x + hv.x, cav.y * h0v.y + hv.y,
                     cav.z * h0v.z + hv.z, cav.w * h0v.w + hv.w};
#pragma unroll
      for (int j = 0; j < 4; ++j) {
        const int c = lane * 4 + j;
        bf16_t* fp = (bf16_t*)&A1F[wv][c >> 5][frag_lane(row, c & 31)];
        fp[frag_elem(c & 31)] = (bf16_t)av[j];
      }
    }
  }
  wave_sync_lds();

  const int mo = (lane < 16) ? 0 : 8;
  const int nn = lane & 15;

  // ---- GEMM1: [16x128] @ [128x256] + b1 -> MF ----
#pragma unroll
  for (int nt = 0; nt < 16; ++nt) {
    v8f acc = zero8();
#pragma unroll
    for (int ks = 0; ks < 4; ++ks) {
      v16bf a = A1F[wv][ks][lane];
      v16bf b = B1[(nt * 4 + ks) * 32 + lane];
      acc = wmma_bf16(a, b, acc);
    }
    const int c = nt * 16 + nn;
    const float bias = nb1[c];
#pragma unroll
    for (int r = 0; r < 8; ++r) MF[wv][mo + r][c] = acc[r] + bias;
  }
  wave_sync_lds();

  // ---- LayerNorm(256) + SiLU -> MBF (fragment layout) ----
  for (int row = 0; row < 16; ++row) {
    float v[8], s = 0.f, ss = 0.f;
#pragma unroll
    for (int j = 0; j < 8; ++j) {
      v[j] = MF[wv][row][lane * 8 + j];
      s += v[j];
      ss += v[j] * v[j];
    }
#pragma unroll
    for (int m = 16; m >= 1; m >>= 1) {
      s  += __shfl_xor(s, m, 32);
      ss += __shfl_xor(ss, m, 32);
    }
    const float mean = s * (1.f / 256.f);
    const float var  = ss * (1.f / 256.f) - mean * mean;
    const float rstd = rsqrtf(var + EPS_);
#pragma unroll
    for (int j = 0; j < 8; ++j) {
      const int c = lane * 8 + j;
      float t = ng1[c] * (v[j] - mean) * rstd + nbe1[c];
      bf16_t* fp = (bf16_t*)&MBF[wv][c >> 5][frag_lane(row, c & 31)];
      fp[frag_elem(c & 31)] = (bf16_t)silu_f(t);
    }
  }
  wave_sync_lds();

  // ---- GEMM2: [16x256] @ [256x128] + b2 ----
  float* MZ = &MF[wv][0][0];  // reuse, viewed with stride 128
  const float av = final_flag ? alp[layer] : 0.f;
#pragma unroll
  for (int nt = 0; nt < 8; ++nt) {
    v8f acc = zero8();
#pragma unroll
    for (int ks = 0; ks < 8; ++ks) {
      v16bf a = MBF[wv][ks][lane];
      v16bf b = B2[(nt * 8 + ks) * 32 + lane];
      acc = wmma_bf16(a, b, acc);
    }
    const int c = nt * 16 + nn;
    const float b2v = nb2[c];
    if (final_flag) {
      if (rc == 16) {
#pragma unroll
        for (int r = 0; r < 8; ++r) {
          const int n = base + mo + r;
          hmut[n * DD_ + c] = hconst[n * DD_ + c] + av * (acc[r] + b2v);
        }
      } else {
#pragma unroll
        for (int r = 0; r < 8; ++r) {
          const int er = mo + r;
          if (er < rc) {
            const int n = base + er;
            hmut[n * DD_ + c] = hconst[n * DD_ + c] + av * (acc[r] + b2v);
          }
        }
      }
    } else {
#pragma unroll
      for (int r = 0; r < 8; ++r) MZ[(mo + r) * DD_ + c] = acc[r] + b2v;
    }
  }

  if (!final_flag) {
    wave_sync_lds();
    // ---- LayerNorm2(128) + SiLU; write zbuf and per-graph sums ----
    for (int row = 0; row < 16; ++row) {
      float v[4], s = 0.f, ss = 0.f;
#pragma unroll
      for (int j = 0; j < 4; ++j) {
        v[j] = MZ[row * DD_ + lane * 4 + j];
        s += v[j];
        ss += v[j] * v[j];
      }
#pragma unroll
      for (int m = 16; m >= 1; m >>= 1) {
        s  += __shfl_xor(s, m, 32);
        ss += __shfl_xor(ss, m, 32);
      }
      const float mean = s * (1.f / 128.f);
      const float var  = ss * (1.f / 128.f) - mean * mean;
      const float rstd = rsqrtf(var + EPS_);
      if (row < rc) {
        const int n = base + row;
        const int g = batch[n];
#pragma unroll
        for (int j = 0; j < 4; ++j) {
          const int c = lane * 4 + j;
          float t = ng2[c] * (v[j] - mean) * rstd + nbe2[c];
          t = silu_f(t);
          zbuf[n * DD_ + c] = t;
          atomicAdd(&gsum[g * DD_ + c], t);
        }
      }
    }
  }
}

// ------------------------------------------------------------------ launch ---

extern "C" void kernel_launch(void* const* d_in, const int* in_sizes, int n_in,
                              void* d_out, int out_size, void* d_ws, size_t ws_size,
                              hipStream_t stream) {
  const float* pos   = (const float*)d_in[0];
  const int*   atype = (const int*)d_in[1];
  const int*   batch = (const int*)d_in[2];
  const int*   eidx  = (const int*)d_in[3];
  const float* emb   = (const float*)d_in[4];
  const float* ew1   = (const float*)d_in[5];
  const float* eb1   = (const float*)d_in[6];
  const float* eg1   = (const float*)d_in[7];
  const float* ebe1  = (const float*)d_in[8];
  const float* ew2   = (const float*)d_in[9];
  const float* eb2   = (const float*)d_in[10];
  const float* ca    = (const float*)d_in[11];
  const float* nw1   = (const float*)d_in[12];
  const float* nb1   = (const float*)d_in[13];
  const float* ng1   = (const float*)d_in[14];
  const float* nbe1  = (const float*)d_in[15];
  const float* nw2   = (const float*)d_in[16];
  const float* nb2   = (const float*)d_in[17];
  const float* ng2   = (const float*)d_in[18];
  const float* nbe2  = (const float*)d_in[19];
  const float* gw    = (const float*)d_in[20];
  const float* gb    = (const float*)d_in[21];
  const float* gms   = (const float*)d_in[22];
  const float* alp   = (const float*)d_in[23];

  const int N = in_sizes[1];
  const int E = in_sizes[3] / 2;
  const int* src  = eidx;
  const int* dstp = eidx + E;

  float* ws    = (float*)d_ws;
  float* h     = ws;
  float* h0    = h     + (size_t)N * DD_;
  float* zbuf  = h0    + (size_t)N * DD_;
  float* gsum  = zbuf  + (size_t)N * DD_;
  float* gvar  = gsum  + NG_ * DD_;
  float* gmean = gvar  + NG_ * DD_;
  float* gvm   = gmean + NG_ * DD_;
  float* psum  = gvm   + NG_ * DD_;
  float* cnts  = psum  + NG_ * DD_;
  float* out   = (float*)d_out;

  // Packed bf16 weight fragments (aligned for v16bf loads).
  uintptr_t pp = (uintptr_t)(cnts + NG_);
  pp = (pp + 255) & ~(uintptr_t)255;
  bf16_t* pw_ew1 = (bf16_t*)pp;
  bf16_t* pw_ew2 = pw_ew1 + (size_t)LL_ * FRAGS_EW1 * 512;
  bf16_t* pw_nw1 = pw_ew2 + (size_t)LL_ * FRAGS_EW2 * 512;
  bf16_t* pw_nw2 = pw_nw1 + (size_t)LL_ * FRAGS_NW1 * 512;

  const int elem_blocks = (int)(((size_t)N * DD_ + 255) / 256);
  const int ewaves  = (E + 15) / 16;
  const int eblocks = (ewaves + EDGE_WAVES - 1) / EDGE_WAVES;
  const int nwaves  = (N + 15) / 16;
  const int nblocks = (nwaves + NODE_WAVES - 1) / NODE_WAVES;

  // One-time weight packing (runs every launch; deterministic).
  for (int i = 0; i < LL_; ++i) {
    pack_b_kernel<<<(FRAGS_EW1 + 7) / 8, 256, 0, stream>>>(
        ew1 + (size_t)i * GG_ * DD_, pw_ew1 + (size_t)i * FRAGS_EW1 * 512,
        GG_, DD_, 2, FRAGS_EW1);
    pack_b_kernel<<<(FRAGS_EW2 + 7) / 8, 256, 0, stream>>>(
        ew2 + (size_t)i * DD_ * DD_, pw_ew2 + (size_t)i * FRAGS_EW2 * 512,
        DD_, DD_, 4, FRAGS_EW2);
    pack_b_kernel<<<(FRAGS_NW1 + 7) / 8, 256, 0, stream>>>(
        nw1 + (size_t)i * DD_ * HH_, pw_nw1 + (size_t)i * FRAGS_NW1 * 512,
        DD_, HH_, 4, FRAGS_NW1);
    pack_b_kernel<<<(FRAGS_NW2 + 7) / 8, 256, 0, stream>>>(
        nw2 + (size_t)i * HH_ * DD_, pw_nw2 + (size_t)i * FRAGS_NW2 * 512,
        HH_, DD_, 8, FRAGS_NW2);
  }

  hipMemsetAsync(cnts, 0, NG_ * sizeof(float), stream);
  count_kernel<<<(N + 255) / 256, 256, 0, stream>>>(batch, cnts, N);
  init_h_kernel<<<elem_blocks, 256, 0, stream>>>(emb, atype, h, N);

  for (int i = 0; i < LL_; ++i) {
    hipMemsetAsync(h0, 0, (size_t)N * DD_ * sizeof(float), stream);
    edge_kernel<<<eblocks, 32 * EDGE_WAVES, 0, stream>>>(
        pos, src, dstp, h, h0,
        pw_ew1 + (size_t)i * FRAGS_EW1 * 512, pw_ew2 + (size_t)i * FRAGS_EW2 * 512,
        eb1 + i * DD_, eg1 + i * DD_, ebe1 + i * DD_, eb2 + i * DD_, E);

    const int fin = (i == LL_ - 1) ? 1 : 0;
    if (!fin) hipMemsetAsync(gsum, 0, NG_ * DD_ * sizeof(float), stream);
    node_kernel<<<nblocks, 32 * NODE_WAVES, 0, stream>>>(
        h, h0, h, ca + i * DD_,
        pw_nw1 + (size_t)i * FRAGS_NW1 * 512, pw_nw2 + (size_t)i * FRAGS_NW2 * 512,
        nb1 + i * HH_, ng1 + i * HH_, nbe1 + i * HH_,
        nb2 + i * DD_, ng2 + i * DD_, nbe2 + i * DD_,
        alp, i, fin, zbuf, gsum, batch, N);

    if (!fin) {
      div_kernel<<<(NG_ * DD_ + 255) / 256, 256, 0, stream>>>(gmean, gsum, cnts);
      hipMemsetAsync(gvar, 0, NG_ * DD_ * sizeof(float), stream);
      center_kernel<<<elem_blocks, 256, 0, stream>>>(zbuf, gmean, gms + i * DD_, batch,
                                                     gvar, N);
      div_kernel<<<(NG_ * DD_ + 255) / 256, 256, 0, stream>>>(gvm, gvar, cnts);
      apply_kernel<<<elem_blocks, 256, 0, stream>>>(h, zbuf, gvm, gw + i * DD_,
                                                    gb + i * DD_, batch, alp, i, N);
    }
  }

  hipMemsetAsync(psum, 0, NG_ * DD_ * sizeof(float), stream);
  pool_accum<<<elem_blocks, 256, 0, stream>>>(h, batch, psum, N);
  pool_finish<<<1, 256, 0, stream>>>(psum, cnts, out);
}
